// RNN_69552700391607
// MI455X (gfx1250) — matrix operand order repaired and checked by reading, not compiled, not run on hardware
//
#include <hip/hip_runtime.h>

#define BB 256
#define TT 128
#define FF 256
#define HH 512
#define CC 10

typedef __attribute__((ext_vector_type(16))) __bf16 v16bf;
typedef __attribute__((ext_vector_type(8)))  float  v8f;

union Frag16 { uint4 u[2]; v16bf v; };

__device__ __forceinline__ unsigned short f32_bf16(float f) {
  unsigned u = __float_as_uint(f);
  u += 0x7FFFu + ((u >> 16) & 1u);          // round-to-nearest-even
  return (unsigned short)(u >> 16);
}
__device__ __forceinline__ unsigned pack_bf16x2(float lo, float hi) {
  return (unsigned)f32_bf16(lo) | ((unsigned)f32_bf16(hi) << 16);
}
// low 32 bits of a generic pointer into LDS == wave-relative LDS byte address
__device__ __forceinline__ unsigned lds_addr32(const void* p) {
  return (unsigned)(size_t)p;
}

// ---------------------------------------------------------------------------
// Pack W [N x K] row-major f32 into WMMA-B-fragment-native bf16 layout.
// B order used here (contiguous k; A matches): lanes 0-15 hold K=0..15, lanes 16-31 K=16..31;
// VGPR d holds K = half*16 + 2d, +1.  Stored lane-contiguous:
// dword = frag*256 + lane*8 + d, so a lane fetches its fragment with 2x b128.
// ---------------------------------------------------------------------------
__global__ void pack_b_kernel(const float* __restrict__ W,
                              unsigned* __restrict__ out, int N, int K) {
  const int KC = K >> 5;
  const int idx = blockIdx.x * 256 + threadIdx.x;
  if (idx >= ((N * K) >> 1)) return;
  const int f    = idx >> 8;
  const int r    = idx & 255;
  const int lane = r >> 3;
  const int d    = r & 7;
  const int nt = f / KC, kc = f - nt * KC;
  const int n  = nt * 16 + (lane & 15);
  const int hf = lane >> 4;
  const int k0 = kc * 32 + hf * 16 + d * 2;
  out[idx] = pack_bf16x2(W[(size_t)n * K + k0], W[(size_t)n * K + k0 + 1]);
}

// ---------------------------------------------------------------------------
// proj = A @ W^T + (b_ih + b_hh).  A: [M x K] (f32 or bf16), W packed.
// Block = 256 threads (8 waves), tile 128(M) x 64(N); wave w owns m-tile w
// and 4 n-tiles.  Each wave stages exactly the 16 LDS rows it consumes
// (sr = tid>>1), so NO barriers are needed: per-wave DS ops are in-order.
// bf16-A staging uses async global->LDS (no VGPR roundtrip), ASYNCcnt fenced.
// ---------------------------------------------------------------------------
template <bool ABF16, int K>
__global__ void __launch_bounds__(256)
gemm_proj(const void* __restrict__ Ap, const unsigned* __restrict__ Bpk,
          const float* __restrict__ bih, const float* __restrict__ bhh,
          float* __restrict__ P) {
  constexpr int KC = K / 32;
  __shared__ unsigned lA[128 * 16];          // 128 rows x 32 bf16 = 8 KB
  const int tid = threadIdx.x, lane = tid & 31, wave = tid >> 5;
  const int mBase = blockIdx.x * 128;
  const int nBase = blockIdx.y * 64;
  const int m = lane & 15, hf = lane >> 4;
  const int sr = tid >> 1, seg = tid & 1;

  const unsigned lA_base = lds_addr32(&lA[0]);
  const unsigned long long a_base = (unsigned long long)Ap;

  v8f acc[4];
  const v8f vzero = {0.f, 0.f, 0.f, 0.f, 0.f, 0.f, 0.f, 0.f};
#pragma unroll
  for (int j = 0; j < 4; ++j) acc[j] = vzero;

  for (int kc = 0; kc < KC; ++kc) {
    // ---- stage this wave's A rows (16 x 32 bf16) into LDS ----
    if constexpr (ABF16) {
      const unsigned dlds = lA_base + (sr * 16 + seg * 8) * 4;
      const unsigned voff = ((unsigned)(mBase + sr) * K + kc * 32 + seg * 16) * 2;
      asm volatile("global_load_async_to_lds_b128 %0, %1, %2"
                   :: "v"(dlds), "v"(voff), "s"(a_base) : "memory");
      asm volatile("global_load_async_to_lds_b128 %0, %1, %2 offset:16"
                   :: "v"(dlds), "v"(voff), "s"(a_base) : "memory");
      asm volatile("s_wait_asynccnt 0x0" ::: "memory");
    } else {
      const unsigned dst = sr * 16 + seg * 8;
      const float4* src = (const float4*)((const float*)Ap +
                          (size_t)(mBase + sr) * K + kc * 32 + seg * 16);
      float4 f0 = src[0], f1 = src[1], f2 = src[2], f3 = src[3];
      lA[dst + 0] = pack_bf16x2(f0.x, f0.y);
      lA[dst + 1] = pack_bf16x2(f0.z, f0.w);
      lA[dst + 2] = pack_bf16x2(f1.x, f1.y);
      lA[dst + 3] = pack_bf16x2(f1.z, f1.w);
      lA[dst + 4] = pack_bf16x2(f2.x, f2.y);
      lA[dst + 5] = pack_bf16x2(f2.z, f2.w);
      lA[dst + 6] = pack_bf16x2(f3.x, f3.y);
      lA[dst + 7] = pack_bf16x2(f3.z, f3.w);
    }

    Frag16 a;
    const uint4* ap = (const uint4*)&lA[(wave * 16 + m) * 16 + hf * 8];
    a.u[0] = ap[0];
    a.u[1] = ap[1];            // +4 dwords

#pragma unroll
    for (int j = 0; j < 4; ++j) {
      const int nt = (nBase >> 4) + j;
      const uint4* bp = (const uint4*)&Bpk[((size_t)nt * KC + kc) * 256 + lane * 8];
      Frag16 b;
      b.u[0] = bp[0];
      b.u[1] = bp[1];
      acc[j] = __builtin_amdgcn_wmma_f32_16x16x32_bf16(
          false, a.v, false, b.v, (short)0, acc[j], false, false);
    }
  }

  // ---- bias + store (C/D layout: n = lane&15, mrow = v + 8*half) ----
#pragma unroll
  for (int j = 0; j < 4; ++j) {
    const int ng = nBase + j * 16 + m;
    const float bias = bih[ng] + bhh[ng];
#pragma unroll
    for (int v = 0; v < 8; ++v) {
      const int mg = mBase + wave * 16 + v + 8 * hf;
      P[(size_t)mg * HH + ng] = acc[j][v] + bias;
    }
  }
}

// ---------------------------------------------------------------------------
// Recurrence: block owns 16 batch rows; h (16x512 bf16) lives in LDS for all
// 128 steps.  Wave w owns n cols [w*64, w*64+64).  Per step: acc = proj[t],
// 16 k-chunks of WMMA (h @ W_hh^T), relu, h written back to LDS, then the
// whole bf16 slice is streamed to global via GLOBAL_STORE_ASYNC_FROM_LDS_B128
// (ASYNCcnt), overlapping the next step's WMMAs.
// Fencing: each wave s_wait_asynccnt(0) BEFORE the barrier, so after the
// barrier all waves' async reads of h are complete and h may be overwritten.
// ---------------------------------------------------------------------------
__global__ void __launch_bounds__(256)
rnn_recur(const float* __restrict__ P, const unsigned* __restrict__ Wpk,
          unsigned short* __restrict__ Hout) {
  __shared__ unsigned hbuf[16 * 256];        // 16 x 512 bf16 = 16 KB
  const int tid = threadIdx.x, lane = tid & 31, wave = tid >> 5;
  const int bBase = blockIdx.x * 16;
  const int m = lane & 15, hf = lane >> 4;

  const unsigned h_base = lds_addr32(&hbuf[0]);
  const unsigned long long hout_base = (unsigned long long)Hout;

  for (int i = tid; i < 16 * 256; i += 256) hbuf[i] = 0u;
  __syncthreads();

  for (int t = 0; t < TT; ++t) {
    // prefetch next step's proj slice while this step computes
    if (t + 1 < TT) {
#pragma unroll
      for (int j = 0; j < 4; ++j) {
        const int ng = wave * 64 + j * 16 + m;
        __builtin_prefetch(&P[((size_t)(bBase + 8 * hf) * TT + t + 1) * HH + ng], 0, 1);
        __builtin_prefetch(&P[((size_t)(bBase + 4 + 8 * hf) * TT + t + 1) * HH + ng], 0, 1);
      }
    }

    v8f acc[4];
#pragma unroll
    for (int j = 0; j < 4; ++j) {
      const int ng = wave * 64 + j * 16 + m;
#pragma unroll
      for (int v = 0; v < 8; ++v) {
        const int mg = v + 8 * hf;
        acc[j][v] = P[((size_t)(bBase + mg) * TT + t) * HH + ng];
      }
    }

    for (int kc = 0; kc < 16; ++kc) {
      Frag16 a;
      const uint4* ap = (const uint4*)&hbuf[m * 256 + kc * 16 + hf * 8];
      a.u[0] = ap[0];
      a.u[1] = ap[1];
#pragma unroll
      for (int j = 0; j < 4; ++j) {
        const int nt = wave * 4 + j;
        const uint4* bp = (const uint4*)&Wpk[((size_t)nt * 16 + kc) * 256 + lane * 8];
        Frag16 b;
        b.u[0] = bp[0];
        b.u[1] = bp[1];
        acc[j] = __builtin_amdgcn_wmma_f32_16x16x32_bf16(
            false, a.v, false, b.v, (short)0, acc[j], false, false);
      }
    }

    // own async stores (step t-1) must be done before anyone rewrites h
    asm volatile("s_wait_asynccnt 0x0" ::: "memory");
    __syncthreads();   // all reads of h done + all waves' async drained

    unsigned short* hb = (unsigned short*)hbuf;
#pragma unroll
    for (int j = 0; j < 4; ++j) {
      const int ng = wave * 64 + j * 16 + m;
#pragma unroll
      for (int v = 0; v < 8; ++v) {
        const int mg = v + 8 * hf;
        float r = acc[j][v];
        r = r > 0.f ? r : 0.f;                        // relu
        hb[mg * 512 + ng] = f32_bf16(r);              // h for next step
      }
    }
    __syncthreads();   // h slice fully written & visible

    // bulk-copy the 16x512 bf16 slice LDS -> Hout[., t, .] asynchronously.
    // wave w streams rows w*2, w*2+1 (1 KB each = 2x b128 per lane).
#pragma unroll
    for (int rr = 0; rr < 2; ++rr) {
      const int r = wave * 2 + rr;
      const unsigned lds = h_base + r * 1024 + lane * 16;
      const unsigned voff =
          (((unsigned)(bBase + r) * TT + (unsigned)t) * HH) * 2 + lane * 16;
      asm volatile("global_store_async_from_lds_b128 %0, %1, %2"
                   :: "v"(voff), "v"(lds), "s"(hout_base) : "memory");
      asm volatile("global_store_async_from_lds_b128 %0, %1, %2 offset:512"
                   :: "v"(voff), "v"(lds), "s"(hout_base) : "memory");
    }
  }
  asm volatile("s_wait_asynccnt 0x0" ::: "memory");
}

// ---------------------------------------------------------------------------
// FC head: out[b, c] = sum_k h1[b, k] * Wfc[c, k] + bfc[c], K = T*H = 65536.
// One block per batch row; vectorized bf16 loads, LDS tree reduction.
// ---------------------------------------------------------------------------
__global__ void __launch_bounds__(256)
fc_kernel(const unsigned short* __restrict__ H1, const float* __restrict__ Wfc,
          const float* __restrict__ bfc, float* __restrict__ out) {
  const int b = blockIdx.x, tid = threadIdx.x;
  const int KK = TT * HH;
  float acc[CC];
#pragma unroll
  for (int c = 0; c < CC; ++c) acc[c] = 0.f;
  const unsigned short* hrow = H1 + (size_t)b * KK;

  for (int k = tid * 8; k < KK; k += 256 * 8) {
    const uint4 hv = *(const uint4*)&hrow[k];
    float h[8];
    h[0] = __uint_as_float(hv.x << 16); h[1] = __uint_as_float(hv.x & 0xFFFF0000u);
    h[2] = __uint_as_float(hv.y << 16); h[3] = __uint_as_float(hv.y & 0xFFFF0000u);
    h[4] = __uint_as_float(hv.z << 16); h[5] = __uint_as_float(hv.z & 0xFFFF0000u);
    h[6] = __uint_as_float(hv.w << 16); h[7] = __uint_as_float(hv.w & 0xFFFF0000u);
#pragma unroll
    for (int c = 0; c < CC; ++c) {
      const float4* wp = (const float4*)&Wfc[(size_t)c * KK + k];
      const float4 a0 = wp[0], a1 = wp[1];
      acc[c] += h[0] * a0.x + h[1] * a0.y + h[2] * a0.z + h[3] * a0.w +
                h[4] * a1.x + h[5] * a1.y + h[6] * a1.z + h[7] * a1.w;
    }
  }

  __shared__ float red[256];
  for (int c = 0; c < CC; ++c) {
    red[tid] = acc[c];
    __syncthreads();
    for (int s = 128; s > 0; s >>= 1) {
      if (tid < s) red[tid] += red[tid + s];
      __syncthreads();
    }
    if (tid == 0) out[b * CC + c] = red[0] + bfc[c];
    __syncthreads();
  }
}

// ---------------------------------------------------------------------------
extern "C" void kernel_launch(void* const* d_in, const int* in_sizes, int n_in,
                              void* d_out, int out_size, void* d_ws, size_t ws_size,
                              hipStream_t stream) {
  (void)in_sizes; (void)n_in; (void)out_size; (void)ws_size;
  const float* x    = (const float*)d_in[0];
  const float* Wih0 = (const float*)d_in[1];
  const float* Whh0 = (const float*)d_in[2];
  const float* bih0 = (const float*)d_in[3];
  const float* bhh0 = (const float*)d_in[4];
  const float* Wih1 = (const float*)d_in[5];
  const float* Whh1 = (const float*)d_in[6];
  const float* bih1 = (const float*)d_in[7];
  const float* bhh1 = (const float*)d_in[8];
  const float* Wfc  = (const float*)d_in[9];
  const float* bfc  = (const float*)d_in[10];
  float* out = (float*)d_out;

  char* ws = (char*)d_ws;
  size_t off = 0;
  float* P           = (float*)(ws + off);          off += (size_t)BB * TT * HH * 4;
  unsigned short* H0 = (unsigned short*)(ws + off); off += (size_t)BB * TT * HH * 2;
  unsigned short* H1 = (unsigned short*)(ws + off); off += (size_t)BB * TT * HH * 2;
  unsigned* Wih0p = (unsigned*)(ws + off);          off += (size_t)HH * FF * 2;
  unsigned* Whh0p = (unsigned*)(ws + off);          off += (size_t)HH * HH * 2;
  unsigned* Wih1p = (unsigned*)(ws + off);          off += (size_t)HH * HH * 2;
  unsigned* Whh1p = (unsigned*)(ws + off);          off += (size_t)HH * HH * 2;

  // 1) pack all weight matrices into WMMA fragment layout (bf16)
  pack_b_kernel<<<(HH * FF / 2) / 256, 256, 0, stream>>>(Wih0, Wih0p, HH, FF);
  pack_b_kernel<<<(HH * HH / 2) / 256, 256, 0, stream>>>(Whh0, Whh0p, HH, HH);
  pack_b_kernel<<<(HH * HH / 2) / 256, 256, 0, stream>>>(Wih1, Wih1p, HH, HH);
  pack_b_kernel<<<(HH * HH / 2) / 256, 256, 0, stream>>>(Whh1, Whh1p, HH, HH);

  // 2) layer 0: input projection + recurrence
  gemm_proj<false, FF><<<dim3(BB * TT / 128, HH / 64), 256, 0, stream>>>(
      x, Wih0p, bih0, bhh0, P);
  rnn_recur<<<BB / 16, 256, 0, stream>>>(P, Whh0p, H0);

  // 3) layer 1: projection from bf16 h0 + recurrence
  gemm_proj<true, HH><<<dim3(BB * TT / 128, HH / 64), 256, 0, stream>>>(
      H0, Wih1p, bih1, bhh1, P);
  rnn_recur<<<BB / 16, 256, 0, stream>>>(P, Whh1p, H1);

  // 4) FC head
  fc_kernel<<<BB, 256, 0, stream>>>(H1, Wfc, bfc, out);
}
